// DigressSTTransformer_17437567221886
// MI455X (gfx1250) — compile-verified, matrix operand order checked by reading, not run on hardware
//
#include <hip/hip_runtime.h>
#include <hip/hip_bf16.h>

typedef __attribute__((ext_vector_type(16))) _Float16 v16h;
typedef __attribute__((ext_vector_type(8)))  _Float16 v8h;
typedef __attribute__((ext_vector_type(8)))  float    v8f;
typedef __attribute__((ext_vector_type(4)))  float    f32x4;
typedef __attribute__((address_space(3)))    _Float16 lds_f16;

// Problem constants (from reference setup_inputs)
constexpr int Bc   = 2;
constexpr int Nn   = 2000;
constexpr int T1c  = 12;
constexpr int Hc   = 128;
constexpr int Lc   = 4;
constexpr int E0c  = 32000;
constexpr int Ec   = E0c + Nn;       // 34000 (self loops appended)
constexpr int Rr   = Bc * Nn * T1c;  // 48000 rows
constexpr int BTc  = Bc * T1c;       // 24
constexpr int EKP  = 288;            // padded K for edge MLP (272 -> 288)

// ---------------------------------------------------------------- helpers
__device__ __forceinline__ float warp_sum(float v) {
#pragma unroll
  for (int off = 16; off > 0; off >>= 1) v += __shfl_xor(v, off, 32);
  return v;
}
__device__ __forceinline__ float warp_max(float v) {
#pragma unroll
  for (int off = 16; off > 0; off >>= 1) v = fmaxf(v, __shfl_xor(v, off, 32));
  return v;
}
__device__ __forceinline__ float gelu_exact(float x) {
  return 0.5f * x * (1.f + erff(x * 0.70710678118654752f));
}
__device__ __forceinline__ unsigned fmono(float x) {
  unsigned u = __float_as_uint(x);
  return (u & 0x80000000u) ? ~u : (u | 0x80000000u);
}
__device__ __forceinline__ float fmono_inv(unsigned u) {
  unsigned v = (u & 0x80000000u) ? (u & 0x7fffffffu) : ~u;
  return __uint_as_float(v);
}

// ---------------------------------------------------------------- weight prep
// W[K,N] f32 -> Wt[Npad,Kpad] f16 (transposed, zero padded).
__global__ void __launch_bounds__(256)
convert_w_kernel(const float* __restrict__ W, _Float16* __restrict__ Wt,
                 int K, int N, int Kpad, int Npad) {
  int i = blockIdx.x * 256 + threadIdx.x;
  if (i >= Npad * Kpad) return;
  int n = i / Kpad;
  int k = i - n * Kpad;
  float v = (n < N && k < K) ? W[(size_t)k * N + n] : 0.f;
  Wt[i] = (_Float16)v;
}

// ---------------------------------------------------------------- WMMA GEMM
// C[M,N] = act(A[M,K] @ W[K,N] + bias), weights pre-transposed f16 Wt[n][k].
// 256 threads = 8 waves; block tile 128(M) x 64(N); 4 wmma accumulators/wave.
// B tile (64 cols x 32 halves) staged via one global_load_async_to_lds_b128
// per thread; A fragments loaded straight from global (two 8-float runs/lane).
__device__ __forceinline__ v16h load_a_frag(const float* __restrict__ A, int M, int K,
                                            int row, int k0, int koff) {
  v16h a;
  if (row < M) {
    const float* pr = A + (size_t)row * K + k0 + koff;
    f32x4 x0 = *(const f32x4*)(pr);
    f32x4 x1 = *(const f32x4*)(pr + 4);
    f32x4 x2 = *(const f32x4*)(pr + 16);
    f32x4 x3 = *(const f32x4*)(pr + 20);
#pragma unroll
    for (int j = 0; j < 4; ++j) {
      a[j]      = (_Float16)x0[j];
      a[4 + j]  = (_Float16)x1[j];
      a[8 + j]  = (_Float16)x2[j];
      a[12 + j] = (_Float16)x3[j];
    }
  } else {
#pragma unroll
    for (int i = 0; i < 16; ++i) a[i] = (_Float16)0.f;
  }
  return a;
}

template <int ACT>
__global__ void __launch_bounds__(256)
gemm_kernel(const float* __restrict__ A, const _Float16* __restrict__ Wt,
            const float* __restrict__ bias, float* __restrict__ C,
            int M, int N, int K, int Kpad) {
  __shared__ _Float16 Bl[64 * 40];  // [col][k] halves, padded stride 40
  const int tid  = threadIdx.x;
  const int lane = tid & 31;
  const int wave = tid >> 5;
  const int m0 = blockIdx.x * 128;
  const int n0 = blockIdx.y * 64;
  const int row  = m0 + wave * 16 + (lane & 15);
  const int koff = (lane < 16) ? 0 : 8;

  // async copy lanes: thread t moves 16B: col bn, halves bp..bp+7
  const int bn = tid >> 2;
  const int bp = (tid & 3) * 8;
  const _Float16* bsrc = Wt + (size_t)(n0 + bn) * Kpad + bp;
  const unsigned blds = (unsigned)(size_t)(lds_f16*)(&Bl[bn * 40 + bp]);

  v8f acc[4];
#pragma unroll
  for (int nt = 0; nt < 4; ++nt)
#pragma unroll
    for (int i = 0; i < 8; ++i) acc[nt][i] = 0.f;

  for (int k0 = 0; k0 < K; k0 += 32) {
    // async stage B chunk (Wt is zero padded -> always in bounds)
    asm volatile("global_load_async_to_lds_b128 %0, %1, off"
                 :: "v"(blds), "v"(bsrc + k0) : "memory");
    v16h afrag = load_a_frag(A, M, K, row, k0, koff);
    asm volatile("s_wait_asynccnt 0x0" ::: "memory");
    __syncthreads();
    v16h bf[4];
#pragma unroll
    for (int nt = 0; nt < 4; ++nt) {
      int nabs = nt * 16 + (lane & 15);
      const v8h* bpp = (const v8h*)(&Bl[nabs * 40 + koff]);
      v8h blo = bpp[0];
      v8h bhi = bpp[2];
#pragma unroll
      for (int i = 0; i < 8; ++i) { bf[nt][i] = blo[i]; bf[nt][8 + i] = bhi[i]; }
    }
#pragma unroll
    for (int nt = 0; nt < 4; ++nt)
      acc[nt] = __builtin_amdgcn_wmma_f32_16x16x32_f16(
          false, afrag, false, bf[nt], (short)0, acc[nt], false, false);
    __syncthreads();
  }

  // epilogue: lane<16 -> rows r..r+7 ; lane>=16 -> rows r+8..r+15
  const int mbase = m0 + wave * 16 + ((lane < 16) ? 0 : 8);
  const int ncol  = n0 + (lane & 15);
#pragma unroll
  for (int nt = 0; nt < 4; ++nt) {
    int n = ncol + nt * 16;
    if (n >= N) continue;
    float bv = bias ? bias[n] : 0.f;
#pragma unroll
    for (int r = 0; r < 8; ++r) {
      int m = mbase + r;
      if (m < M) {
        float v = acc[nt][r] + bv;
        if (ACT == 1) v = gelu_exact(v);
        C[(size_t)m * N + n] = v;
      }
    }
  }
}

// ---------------------------------------------------------------- small kernels
__global__ void __launch_bounds__(256)
build_edges_kernel(const int* __restrict__ ei, int* __restrict__ dst,
                   int* __restrict__ src, float* __restrict__ deg) {
  int i = blockIdx.x * 256 + threadIdx.x;
  if (i >= Ec) return;
  int d, s;
  if (i < E0c) { d = ei[i]; s = ei[E0c + i]; }
  else         { d = s = i - E0c; }
  dst[i] = d; src[i] = s;
  atomicAdd(&deg[d], 1.0f);
}

__global__ void __launch_bounds__(256)
degmax_kernel(const float* __restrict__ deg, float* __restrict__ dm) {
  __shared__ float red[8];
  float m = 1.0f;
  for (int i = threadIdx.x; i < Nn; i += 256) m = fmaxf(m, deg[i]);
  m = warp_max(m);
  if ((threadIdx.x & 31) == 0) red[threadIdx.x >> 5] = m;
  __syncthreads();
  if (threadIdx.x == 0) {
    float r = red[0];
#pragma unroll
    for (int i = 1; i < 8; ++i) r = fmaxf(r, red[i]);
    dm[0] = r;
  }
}

__global__ void __launch_bounds__(256)
e_init_kernel(const float* __restrict__ deg, const float* __restrict__ dm,
              const int* __restrict__ dst, const int* __restrict__ src,
              const float* __restrict__ We, const float* __restrict__ be,
              float* __restrict__ e) {
  int i = blockIdx.x * 256 + threadIdx.x;
  if (i >= Ec * 16) return;
  int ei = i >> 4, c = i & 15;
  float dd = deg[dst[ei]];
  float ds = deg[src[ei]];
  float dmax = dm[0];
  float f0 = 1.f / fmaxf(dd, 1.f);
  float f1 = ds / dmax;
  float f2 = dd / dmax;
  e[i] = f0 * We[c] + f1 * We[16 + c] + f2 * We[32 + c] + be[c];
}

__global__ void __launch_bounds__(256)
tre_kernel(const float* __restrict__ temb, const float* __restrict__ Wtr,
           const float* __restrict__ btr, float* __restrict__ tre) {
  int i = blockIdx.x * 256 + threadIdx.x;
  if (i >= T1c * Hc) return;
  int t = i >> 7, c = i & 127;
  float s = btr[c];
#pragma unroll
  for (int j = 0; j < 16; ++j) s += temb[t * 16 + j] * Wtr[j * 128 + c];
  tre[i] = s;
}

// input embedding: x28 = [X_k, t_emb] -> @W_in+b_in -> LN -> + step_embed
__global__ void __launch_bounds__(256)
embed_kernel(const float* __restrict__ Xk, const float* __restrict__ temb,
             const float* __restrict__ Win, const float* __restrict__ bin,
             const float* __restrict__ g, const float* __restrict__ be,
             const float* __restrict__ stepe, const int* __restrict__ kidx,
             float* __restrict__ h) {
  int gw = blockIdx.x * 8 + (threadIdx.x >> 5);  // row id
  int lane = threadIdx.x & 31;
  int b = gw / (Nn * T1c);
  int t = gw % T1c;
  float x[28];
  const float* xp = Xk + (size_t)gw * 12;
#pragma unroll
  for (int i = 0; i < 12; ++i) x[i] = xp[i];
#pragma unroll
  for (int i = 0; i < 16; ++i) x[12 + i] = temb[t * 16 + i];
  float v[4];
#pragma unroll
  for (int j = 0; j < 4; ++j) {
    int c = lane + 32 * j;
    float a = bin[c];
#pragma unroll
    for (int k = 0; k < 28; ++k) a += x[k] * Win[k * 128 + c];
    v[j] = a;
  }
  float s = warp_sum(v[0] + v[1] + v[2] + v[3]);
  float mean = s * (1.f / 128.f);
  float q = 0.f;
#pragma unroll
  for (int j = 0; j < 4; ++j) { float d = v[j] - mean; q += d * d; }
  q = warp_sum(q);
  float rs = rsqrtf(q * (1.f / 128.f) + 1e-5f);
  const float* se = stepe + (size_t)kidx[b] * 128;
#pragma unroll
  for (int j = 0; j < 4; ++j) {
    int c = lane + 32 * j;
    h[(size_t)gw * 128 + c] = (v[j] - mean) * rs * g[c] + be[c] + se[c];
  }
}

// temporal MHA over T1=12, one wave per (b,n,head)
__global__ void __launch_bounds__(256)
temporal_attn_kernel(const float* __restrict__ qkv, float* __restrict__ out) {
  int gw = blockIdx.x * 8 + (threadIdx.x >> 5);  // 0..B*N*NH-1
  int lane = threadIdx.x & 31;
  int head = gw & 3;
  int bn = gw >> 2;
  int col = head * 32 + lane;
  float q[12], k[12], v[12];
#pragma unroll
  for (int t = 0; t < 12; ++t) {
    const float* p = qkv + (size_t)(bn * 12 + t) * 384;
    q[t] = p[col];
    k[t] = p[128 + col];
    v[t] = p[256 + col];
  }
  const float scale = 0.17677669529663687f;  // 1/sqrt(32)
#pragma unroll
  for (int qt = 0; qt < 12; ++qt) {
    float s[12];
#pragma unroll
    for (int kt = 0; kt < 12; ++kt) s[kt] = warp_sum(q[qt] * k[kt]) * scale;
    float mx = s[0];
#pragma unroll
    for (int kt = 1; kt < 12; ++kt) mx = fmaxf(mx, s[kt]);
    float sum = 0.f, o = 0.f;
#pragma unroll
    for (int kt = 0; kt < 12; ++kt) {
      float p = expf(s[kt] - mx);
      sum += p;
      o += p * v[kt];
    }
    out[(size_t)(bn * 12 + qt) * 128 + col] = o / sum;
  }
}

// h = LN(h [+ x]) * g + b, H=128, one wave per row
template <bool ADD>
__global__ void __launch_bounds__(256)
ln128_kernel(float* __restrict__ h, const float* __restrict__ x,
             const float* __restrict__ g, const float* __restrict__ b, int rows) {
  int gw = blockIdx.x * 8 + (threadIdx.x >> 5);
  if (gw >= rows) return;
  int lane = threadIdx.x & 31;
  float* hp = h + (size_t)gw * 128;
  float v[4];
#pragma unroll
  for (int j = 0; j < 4; ++j) {
    float t = hp[lane + 32 * j];
    if (ADD) t += x[(size_t)gw * 128 + lane + 32 * j];
    v[j] = t;
  }
  float s = warp_sum(v[0] + v[1] + v[2] + v[3]);
  float mean = s * (1.f / 128.f);
  float q = 0.f;
#pragma unroll
  for (int j = 0; j < 4; ++j) { float d = v[j] - mean; q += d * d; }
  q = warp_sum(q);
  float rs = rsqrtf(q * (1.f / 128.f) + 1e-5f);
#pragma unroll
  for (int j = 0; j < 4; ++j) {
    int c = lane + 32 * j;
    hp[c] = (v[j] - mean) * rs * g[c] + b[c];
  }
}

__global__ void __launch_bounds__(256)
add_trow_kernel(float* __restrict__ h, const float* __restrict__ tre) {
  int i = blockIdx.x * 256 + threadIdx.x;
  int r = i >> 7, c = i & 127;
  int t = r % T1c;
  h[i] += tre[t * 128 + c];
}

__global__ void __launch_bounds__(256)
resadd_kernel(float* __restrict__ h, const float* __restrict__ p) {
  int i = blockIdx.x * 256 + threadIdx.x;
  h[i] += p[i];
}

__global__ void __launch_bounds__(256)
edge_mod_kernel(const float* __restrict__ eb, const float* __restrict__ Wem,
                const float* __restrict__ bem, const float* __restrict__ Wea,
                const float* __restrict__ bea, float* __restrict__ multb,
                float* __restrict__ addb) {
  int i = blockIdx.x * 256 + threadIdx.x;
  if (i >= Ec * 4) return;
  int e = i >> 2, hh = i & 3;
  float m = bem[hh], a = bea[hh];
#pragma unroll
  for (int j = 0; j < 16; ++j) {
    float ev = eb[e * 16 + j];
    m += ev * Wem[j * 4 + hh];
    a += ev * Wea[j * 4 + hh];
  }
  multb[i] = m + 1.f;
  addb[i] = a;
}

// one wave per (bt, e): 4 head dot-products + global max
__global__ void __launch_bounds__(256)
scores_kernel(const float* __restrict__ Q, const float* __restrict__ Kb,
              const int* __restrict__ dst, const int* __restrict__ src,
              const float* __restrict__ multb, const float* __restrict__ addb,
              float* __restrict__ sc, unsigned* __restrict__ smax) {
  int gw = blockIdx.x * 8 + (threadIdx.x >> 5);
  if (gw >= BTc * Ec) return;
  int lane = threadIdx.x & 31;
  int bt = gw / Ec;
  int e = gw - bt * Ec;
  int b = bt / T1c;
  int t = bt - b * T1c;
  size_t qrow = ((size_t)(b * Nn + dst[e]) * T1c + t) * 128;
  size_t krow = ((size_t)(b * Nn + src[e]) * T1c + t) * 128;
  const float scl = 0.17677669529663687f;
#pragma unroll
  for (int hh = 0; hh < 4; ++hh) {
    float d = warp_sum(Q[qrow + hh * 32 + lane] * Kb[krow + hh * 32 + lane]);
    float s = d * scl * multb[e * 4 + hh] + addb[e * 4 + hh];
    if (lane == 0) {
      sc[((size_t)bt * Ec + e) * 4 + hh] = s;
      atomicMax(&smax[bt * 4 + hh], fmono(s));
    }
  }
}

__global__ void __launch_bounds__(256)
expdenom_kernel(float* __restrict__ sc, const unsigned* __restrict__ smax,
                const int* __restrict__ dst, float* __restrict__ denom) {
  int i = blockIdx.x * 256 + threadIdx.x;
  if (i >= BTc * Ec * 4) return;
  int hh = i & 3;
  int te = i >> 2;
  int bt = te / Ec;
  int e = te - bt * Ec;
  float mx = fmono_inv(smax[bt * 4 + hh]);
  float ex = expf(sc[i] - mx);
  sc[i] = ex;
  atomicAdd(&denom[((size_t)bt * Nn + dst[e]) * 4 + hh], ex);
}

// one wave per (bt, e): scatter attn * V[src] into out[dst]
__global__ void __launch_bounds__(256)
agg_kernel(const float* __restrict__ sc, const float* __restrict__ denom,
           const int* __restrict__ dst, const int* __restrict__ src,
           const float* __restrict__ V, float* __restrict__ outp) {
  int gw = blockIdx.x * 8 + (threadIdx.x >> 5);
  if (gw >= BTc * Ec) return;
  int lane = threadIdx.x & 31;
  int bt = gw / Ec;
  int e = gw - bt * Ec;
  int b = bt / T1c;
  int t = bt - b * T1c;
  int dn = dst[e], sn = src[e];
  size_t orow = ((size_t)(b * Nn + dn) * T1c + t) * 128;
  size_t vrow = ((size_t)(b * Nn + sn) * T1c + t) * 128;
  float a[4];
#pragma unroll
  for (int hh = 0; hh < 4; ++hh)
    a[hh] = sc[((size_t)bt * Ec + e) * 4 + hh] /
            (denom[((size_t)bt * Nn + dn) * 4 + hh] + 1e-9f);
#pragma unroll
  for (int hh = 0; hh < 4; ++hh) {
    int d = hh * 32 + lane;
    atomicAdd(&outp[orow + d], a[hh] * V[vrow + d]);
  }
}

__global__ void __launch_bounds__(256)
hmean_kernel(const float* __restrict__ h, float* __restrict__ hm) {
  int i = blockIdx.x * 256 + threadIdx.x;  // n*128 + c
  int n = i >> 7, c = i & 127;
  float s = 0.f;
#pragma unroll
  for (int b = 0; b < Bc; ++b)
#pragma unroll
    for (int t = 0; t < T1c; ++t)
      s += h[((size_t)(b * Nn + n) * T1c + t) * 128 + c];
  hm[i] = s * (1.f / (float)BTc);
}

// e_in row stride 288: [h_mean[dst] | h_mean[src] | e | 16 zero pad cols]
__global__ void __launch_bounds__(256)
concat_kernel(const float* __restrict__ hm, const int* __restrict__ dst,
              const int* __restrict__ src, const float* __restrict__ eb,
              float* __restrict__ ein) {
  int i = blockIdx.x * 256 + threadIdx.x;
  if (i >= Ec * EKP) return;
  int e = i / EKP;
  int c = i - e * EKP;
  float v = 0.f;
  if (c < 128)      v = hm[(size_t)dst[e] * 128 + c];
  else if (c < 256) v = hm[(size_t)src[e] * 128 + (c - 128)];
  else if (c < 272) v = eb[(size_t)e * 16 + (c - 256)];
  ein[i] = v;
}

// e = LN(e + e_new) over 16, one 16-lane group per row (2 rows per wave)
__global__ void __launch_bounds__(256)
ln16_kernel(float* __restrict__ e, const float* __restrict__ em,
            const float* __restrict__ g, const float* __restrict__ b) {
  int row = blockIdx.x * 16 + (threadIdx.x >> 4);
  int l = threadIdx.x & 15;
  float v = e[(size_t)row * 16 + l] + em[(size_t)row * 16 + l];
  float s = v;
#pragma unroll
  for (int off = 8; off > 0; off >>= 1) s += __shfl_xor(s, off, 32);
  float mean = s * (1.f / 16.f);
  float d = v - mean;
  float q = d * d;
#pragma unroll
  for (int off = 8; off > 0; off >>= 1) q += __shfl_xor(q, off, 32);
  float rs = rsqrtf(q * (1.f / 16.f) + 1e-5f);
  e[(size_t)row * 16 + l] = d * rs * g[l] + b[l];
}

// softmax(h @ W_out + b_out) over 4 classes, one wave per row
__global__ void __launch_bounds__(256)
out_kernel(const float* __restrict__ h, const float* __restrict__ Wout,
           const float* __restrict__ bout, float* __restrict__ out) {
  int gw = blockIdx.x * 8 + (threadIdx.x >> 5);
  int lane = threadIdx.x & 31;
  const float* hp = h + (size_t)gw * 128;
  float hv[4];
#pragma unroll
  for (int j = 0; j < 4; ++j) hv[j] = hp[lane + 32 * j];
  float lg[4];
#pragma unroll
  for (int o = 0; o < 4; ++o) {
    float s = 0.f;
#pragma unroll
    for (int j = 0; j < 4; ++j) s += hv[j] * Wout[(lane + 32 * j) * 4 + o];
    lg[o] = warp_sum(s) + bout[o];
  }
  float mx = fmaxf(fmaxf(lg[0], lg[1]), fmaxf(lg[2], lg[3]));
  float p0 = expf(lg[0] - mx), p1 = expf(lg[1] - mx);
  float p2 = expf(lg[2] - mx), p3 = expf(lg[3] - mx);
  float inv = 1.f / (p0 + p1 + p2 + p3);
  if (lane < 4) {
    float v = (lane == 0) ? p0 : (lane == 1) ? p1 : (lane == 2) ? p2 : p3;
    out[(size_t)gw * 4 + lane] = v * inv;
  }
}

// ---------------------------------------------------------------- launch
struct WtDesc { _Float16* wt; int kpad; };

extern "C" void kernel_launch(void* const* d_in, const int* in_sizes, int n_in,
                              void* d_out, int out_size, void* d_ws, size_t ws_size,
                              hipStream_t stream) {
  (void)in_sizes; (void)n_in; (void)out_size; (void)ws_size;
  const float* X_k      = (const float*)d_in[0];
  const float* time_emb = (const float*)d_in[1];
  const float* step_emb = (const float*)d_in[2];
  const float* W_tr     = (const float*)d_in[3];
  const float* b_tr     = (const float*)d_in[4];
  const float* W_in     = (const float*)d_in[5];
  const float* b_in     = (const float*)d_in[6];
  const float* g_in     = (const float*)d_in[7];
  const float* be_in    = (const float*)d_in[8];
  const float* W_e      = (const float*)d_in[9];
  const float* b_e      = (const float*)d_in[10];
  const float* aw  = (const float*)d_in[11];
  const float* ab  = (const float*)d_in[12];
  const float* ow  = (const float*)d_in[13];
  const float* ob  = (const float*)d_in[14];
  const float* tng = (const float*)d_in[15];
  const float* tnb = (const float*)d_in[16];
  const float* f1w = (const float*)d_in[17];
  const float* f1b = (const float*)d_in[18];
  const float* f2w = (const float*)d_in[19];
  const float* f2b = (const float*)d_in[20];
  const float* fng = (const float*)d_in[21];
  const float* fnb = (const float*)d_in[22];
  const float* Wq  = (const float*)d_in[23];
  const float* Wk  = (const float*)d_in[24];
  const float* Wv  = (const float*)d_in[25];
  const float* Wo  = (const float*)d_in[26];
  const float* bo  = (const float*)d_in[27];
  const float* Wem = (const float*)d_in[28];
  const float* bem = (const float*)d_in[29];
  const float* Wea = (const float*)d_in[30];
  const float* bea = (const float*)d_in[31];
  const float* e1w = (const float*)d_in[32];
  const float* e1b = (const float*)d_in[33];
  const float* e2w = (const float*)d_in[34];
  const float* e2b = (const float*)d_in[35];
  const float* gng = (const float*)d_in[36];
  const float* gnb = (const float*)d_in[37];
  const float* eng = (const float*)d_in[38];
  const float* enb = (const float*)d_in[39];
  const float* W_out = (const float*)d_in[40];
  const float* b_out = (const float*)d_in[41];
  const int* k_index    = (const int*)d_in[42];
  const int* edge_index = (const int*)d_in[43];
  float* out = (float*)d_out;

  // workspace carve-out (256B aligned regions)
  char* p = (char*)d_ws;
  auto carve = [&](size_t bytes) -> char* {
    char* r = p;
    p += (bytes + 255) & ~(size_t)255;
    return r;
  };
  float* h      = (float*)carve((size_t)Rr * 128 * 4);
  float* tmp1   = (float*)carve((size_t)Rr * 384 * 4);
  float* tmp2   = (float*)carve((size_t)Rr * 128 * 4);
  float* e_buf  = (float*)carve((size_t)Ec * 16 * 4);
  float* e_in   = (float*)carve((size_t)Ec * EKP * 4);
  float* e_mid  = (float*)carve((size_t)Ec * 64 * 4);
  float* e_new  = (float*)carve((size_t)Ec * 16 * 4);
  float* deg    = (float*)carve((size_t)Nn * 4);
  float* degmx  = (float*)carve(256);
  int*   dstA   = (int*)carve((size_t)Ec * 4);
  int*   srcA   = (int*)carve((size_t)Ec * 4);
  float* scores = (float*)carve((size_t)BTc * Ec * 4 * 4);
  unsigned* smax = (unsigned*)carve((size_t)BTc * 4 * 4);
  float* denom  = (float*)carve((size_t)BTc * Nn * 4 * 4);
  float* multb  = (float*)carve((size_t)Ec * 4 * 4);
  float* addb   = (float*)carve((size_t)Ec * 4 * 4);
  float* hmean  = (float*)carve((size_t)Nn * 128 * 4);
  float* tre    = (float*)carve((size_t)T1c * 128 * 4);
  _Float16* wpool = (_Float16*)carve((size_t)4 * 1024 * 1024);  // f16 weight pool
  size_t woff = 0;

  // pre-convert a weight matrix to transposed/padded f16
  auto prep = [&](const float* W, int K, int N) -> WtDesc {
    int Kpad = (K + 31) & ~31;
    int Npad = (N + 63) & ~63;
    _Float16* wt = wpool + woff;
    woff += (size_t)Kpad * Npad;
    int total = Kpad * Npad;
    convert_w_kernel<<<(total + 255) / 256, 256, 0, stream>>>(W, wt, K, N, Kpad, Npad);
    return WtDesc{wt, Kpad};
  };

  WtDesc Waw[Lc], Wow[Lc], Wf1[Lc], Wf2[Lc], Wwq[Lc], Wwk[Lc], Wwv[Lc], Wwo[Lc],
         We1[Lc], We2[Lc];
  for (int l = 0; l < Lc; ++l) {
    Waw[l] = prep(aw  + (size_t)l * 128 * 384, 128, 384);
    Wow[l] = prep(ow  + (size_t)l * 128 * 128, 128, 128);
    Wf1[l] = prep(f1w + (size_t)l * 128 * 256, 128, 256);
    Wf2[l] = prep(f2w + (size_t)l * 256 * 128, 256, 128);
    Wwq[l] = prep(Wq  + (size_t)l * 128 * 128, 128, 128);
    Wwk[l] = prep(Wk  + (size_t)l * 128 * 128, 128, 128);
    Wwv[l] = prep(Wv  + (size_t)l * 128 * 128, 128, 128);
    Wwo[l] = prep(Wo  + (size_t)l * 128 * 128, 128, 128);
    We1[l] = prep(e1w + (size_t)l * 272 * 64, 272, 64);    // Kpad = 288
    We2[l] = prep(e2w + (size_t)l * 64 * 16, 64, 16);
  }

  auto gemm = [&](const float* A, const WtDesc& w, const float* bias, float* C,
                  int M, int Ncols, int K, int act) {
    dim3 g((M + 127) / 128, (Ncols + 63) / 64);
    if (act) gemm_kernel<1><<<g, 256, 0, stream>>>(A, w.wt, bias, C, M, Ncols, K, w.kpad);
    else     gemm_kernel<0><<<g, 256, 0, stream>>>(A, w.wt, bias, C, M, Ncols, K, w.kpad);
  };

  // ---- graph prep + input embedding ----
  hipMemsetAsync(deg, 0, (size_t)Nn * 4, stream);
  build_edges_kernel<<<(Ec + 255) / 256, 256, 0, stream>>>(edge_index, dstA, srcA, deg);
  degmax_kernel<<<1, 256, 0, stream>>>(deg, degmx);
  e_init_kernel<<<(Ec * 16 + 255) / 256, 256, 0, stream>>>(deg, degmx, dstA, srcA, W_e, b_e, e_buf);
  tre_kernel<<<(T1c * Hc + 255) / 256, 256, 0, stream>>>(time_emb, W_tr, b_tr, tre);
  embed_kernel<<<Rr / 8, 256, 0, stream>>>(X_k, time_emb, W_in, b_in, g_in, be_in,
                                           step_emb, k_index, h);

  for (int l = 0; l < Lc; ++l) {
    // temporal attention
    gemm(h, Waw[l], ab + (size_t)l * 384, tmp1, Rr, 384, 128, 0);   // QKV
    temporal_attn_kernel<<<Rr * 4 / (12 * 8), 256, 0, stream>>>(tmp1, tmp2);
    gemm(tmp2, Wow[l], ob + (size_t)l * 128, tmp1, Rr, 128, 128, 0);
    ln128_kernel<true><<<Rr / 8, 256, 0, stream>>>(h, tmp1, tng + l * 128, tnb + l * 128, Rr);
    // FFN
    gemm(h, Wf1[l], f1b + (size_t)l * 256, tmp1, Rr, 256, 128, 1);  // GELU
    gemm(tmp1, Wf2[l], f2b + (size_t)l * 128, tmp2, Rr, 128, 256, 0);
    ln128_kernel<true><<<Rr / 8, 256, 0, stream>>>(h, tmp2, fng + l * 128, fnb + l * 128, Rr);
    add_trow_kernel<<<Rr * 128 / 256, 256, 0, stream>>>(h, tre);

    // sparse graph attention
    float* Qb = tmp1;
    float* Kb = tmp1 + (size_t)Rr * 128;
    float* Vb = tmp1 + (size_t)2 * Rr * 128;
    gemm(h, Wwq[l], nullptr, Qb, Rr, 128, 128, 0);
    gemm(h, Wwk[l], nullptr, Kb, Rr, 128, 128, 0);
    gemm(h, Wwv[l], nullptr, Vb, Rr, 128, 128, 0);
    edge_mod_kernel<<<(Ec * 4 + 255) / 256, 256, 0, stream>>>(
        e_buf, Wem + (size_t)l * 64, bem + (size_t)l * 4,
        Wea + (size_t)l * 64, bea + (size_t)l * 4, multb, addb);
    hipMemsetAsync(smax, 0, (size_t)BTc * 4 * 4, stream);
    scores_kernel<<<BTc * Ec / 8, 256, 0, stream>>>(Qb, Kb, dstA, srcA, multb, addb,
                                                    scores, smax);
    hipMemsetAsync(denom, 0, (size_t)BTc * Nn * 4 * 4, stream);
    expdenom_kernel<<<BTc * Ec * 4 / 256, 256, 0, stream>>>(scores, smax, dstA, denom);
    hipMemsetAsync(tmp2, 0, (size_t)Rr * 128 * 4, stream);
    agg_kernel<<<BTc * Ec / 8, 256, 0, stream>>>(scores, denom, dstA, srcA, Vb, tmp2);
    gemm(tmp2, Wwo[l], bo + (size_t)l * 128, tmp1, Rr, 128, 128, 0);
    resadd_kernel<<<Rr * 128 / 256, 256, 0, stream>>>(h, tmp1);

    // edge update
    hmean_kernel<<<Nn * 128 / 256, 256, 0, stream>>>(h, hmean);
    concat_kernel<<<(Ec * EKP + 255) / 256, 256, 0, stream>>>(hmean, dstA, srcA, e_buf, e_in);
    gemm(e_in, We1[l], e1b + (size_t)l * 64, e_mid, Ec, 64, EKP, 1);  // GELU, K padded to 288
    gemm(e_mid, We2[l], e2b + (size_t)l * 16, e_new, Ec, 16, 64, 0);
    ln16_kernel<<<Ec / 16, 256, 0, stream>>>(e_buf, e_new, eng + l * 16, enb + l * 16);

    // node LN (no residual)
    ln128_kernel<false><<<Rr / 8, 256, 0, stream>>>(h, nullptr, gng + l * 128, gnb + l * 128, Rr);
  }

  out_kernel<<<Rr / 8, 256, 0, stream>>>(h, W_out, b_out, out);
}
